// MoETransformer_66829691126407
// MI455X (gfx1250) — matrix operand order repaired and checked by reading, not compiled, not run on hardware
//
#include <hip/hip_runtime.h>

// ---------------- constants ----------------
constexpr int TOK   = 1024;   // b*t
constexpr int SEQ   = 512;
constexpr int BATCH = 2;
constexpr int DIM   = 768;
constexpr int NHEAD = 12;
constexpr int DHEAD = 64;
constexpr int HID   = 3072;
constexpr int NEXP  = 8;
constexpr int VOCAB = 50257;
constexpr int NLAYER = 4;

// ---------------- WMMA / TDM types ----------------
typedef __attribute__((ext_vector_type(16))) __bf16 v16bf;
typedef __attribute__((ext_vector_type(8)))  float  v8f;
typedef __attribute__((ext_vector_type(4)))  unsigned int v4u;
typedef __attribute__((ext_vector_type(8)))  int v8i;
typedef __attribute__((ext_vector_type(4)))  int v4i;

union BF16Frag {
  v16bf v;
  uint4 q[2];
  unsigned int w[8];
  unsigned short u[16];
};

// native f32 -> bf16 conversions (backend emits v_cvt_pk_bf16_f32)
__device__ __forceinline__ unsigned int pk2bf(float a, float b) {
  union { __bf16 h[2]; unsigned int u; } t;
  t.h[0] = (__bf16)a; t.h[1] = (__bf16)b;
  return t.u;
}
__device__ __forceinline__ unsigned short f2bfb(float a) {
  union { __bf16 h; unsigned short s; } t;
  t.h = (__bf16)a;
  return t.s;
}

__device__ __forceinline__ v8f wmma_bf16(const BF16Frag& a, const BF16Frag& b, v8f c) {
  return __builtin_amdgcn_wmma_f32_16x16x32_bf16(false, a.v, false, b.v,
                                                 (short)0, c, false, false);
}

__device__ __forceinline__ float gelu_exact(float x) {
  return 0.5f * x * (1.0f + erff(x * 0.7071067811865476f));
}

// ---------------- TDM: async 2-D fp32 tile load, global -> LDS ----------------
// Loads tile (tile_dim0=32 cols x tileRows rows) of a row-major fp32 matrix whose
// remaining extent from the tile origin is remCols x remRows (OOB rows read as 0).
// LDS padding: +4 DWORDs after every 32 DWORDs -> row stride 36 floats (144 B).
__device__ __forceinline__ void tdm_load_tile(void* ldsDst, const void* gsrc,
                                              int tileRows, int remCols, int remRows,
                                              int strideElems) {
  unsigned long long ga = (unsigned long long)(uintptr_t)gsrc;
  unsigned lds = (unsigned)(uintptr_t)ldsDst;
  v4u g0;
  g0[0] = 1u;                                                  // count=1 valid descriptor
  g0[1] = lds;                                                 // lds_addr
  g0[2] = (unsigned)(ga & 0xFFFFFFFFu);                        // global_addr[31:0]
  g0[3] = (unsigned)((ga >> 32) & 0x01FFFFFFu) | (2u << 30);   // global_addr[56:32] | type=2
  v8i g1;
  g1[0] = (int)((2u << 16) | (1u << 20) | (4u << 22) | (3u << 25)); // 4B elems, pad_en, 32dw + 4dw
  g1[1] = (int)(((unsigned)remCols & 0xFFFFu) << 16);               // tensor_dim0[15:0]
  g1[2] = (int)((((unsigned)remCols >> 16) & 0xFFFFu) |
                (((unsigned)remRows & 0xFFFFu) << 16));             // dim0[31:16] | dim1[15:0]
  g1[3] = (int)((((unsigned)remRows >> 16) & 0xFFFFu) | (32u << 16)); // dim1[31:16] | tile_dim0=32
  g1[4] = (int)((unsigned)tileRows & 0xFFFFu);                      // tile_dim1 (tile_dim2=0)
  g1[5] = (int)(unsigned)strideElems;                               // tensor_dim0_stride[31:0]
  g1[6] = 0;
  g1[7] = 0;
  v4i z4 = {0, 0, 0, 0};
#if defined(__clang_major__) && (__clang_major__ >= 23)
  v8i z8 = {0, 0, 0, 0, 0, 0, 0, 0};
  __builtin_amdgcn_tensor_load_to_lds(g0, g1, z4, z4, z8, 0);
#else
  __builtin_amdgcn_tensor_load_to_lds(g0, g1, z4, z4, 0);
#endif
}

// ---------------- embedding + sinusoidal PE ----------------
__global__ void embed_kernel(const int* __restrict__ x, const float* __restrict__ emb,
                             float* __restrict__ h) {
  const int tok = blockIdx.x;
  const int tq  = tok & (SEQ - 1);
  const int xi  = x[tok];
  #pragma unroll
  for (int c = 0; c < 3; ++c) {
    int d = c * 256 + threadIdx.x;
    float ang = (float)tq * __expf(-(float)(d & ~1) * (9.210340371976184f / (float)DIM));
    float pe  = (d & 1) ? cosf(ang) : sinf(ang);
    h[(size_t)tok * DIM + d] = emb[(size_t)xi * DIM + d] + pe;
  }
}

// ---------------- layernorm (one block per token) ----------------
__global__ void __launch_bounds__(256) ln_kernel(const float* __restrict__ x,
                                                 const float* __restrict__ g,
                                                 const float* __restrict__ b,
                                                 float* __restrict__ y) {
  const int tok = blockIdx.x;
  const int tid = threadIdx.x;
  const int lane = tid & 31, wid = tid >> 5;
  __shared__ float s1[8], s2[8];

  float vals[3]; float s = 0.f, sq = 0.f;
  #pragma unroll
  for (int c = 0; c < 3; ++c) {
    float v = x[(size_t)tok * DIM + c * 256 + tid];
    vals[c] = v; s += v; sq += v * v;
  }
  #pragma unroll
  for (int off = 16; off; off >>= 1) { s += __shfl_xor(s, off); sq += __shfl_xor(sq, off); }
  if (lane == 0) { s1[wid] = s; s2[wid] = sq; }
  __syncthreads();
  float ts = 0.f, tsq = 0.f;
  #pragma unroll
  for (int j = 0; j < 8; ++j) { ts += s1[j]; tsq += s2[j]; }
  float mean = ts * (1.0f / DIM);
  float var  = tsq * (1.0f / DIM) - mean * mean;
  float rs   = rsqrtf(var + 1e-5f);
  #pragma unroll
  for (int c = 0; c < 3; ++c) {
    int d = c * 256 + tid;
    y[(size_t)tok * DIM + d] = (vals[c] - mean) * rs * g[d] + b[d];
  }
}

// ---------------- generic bf16 WMMA GEMM with TDM-staged tiles ----------------
// C = [C +] [rscale*](A W^T + bias), optional GELU, optional fused QKV head split.
// A:[M,K] lda, W:[N,K] ldw, C:[M,N] ldc.  Block tile 128x64, 8 waves of 32x32.
// fp32 tiles stream to LDS via tensor_load_to_lds (2-deep pipeline, TENSORcnt).
template<int BIAS, int GELU, int ACC, int RSC, int SPLITQ>
__global__ void __launch_bounds__(256) gemm_bf16_kernel(
    const float* __restrict__ A, int lda,
    const float* __restrict__ W, int ldw,
    const float* __restrict__ bias,
    const float* __restrict__ rscale, int rs_stride,
    float* __restrict__ C, int ldc,
    int M, int N, int K) {
  constexpr int BM = 128, BN = 64, BK = 32, LW = 36;   // 36 floats = 144 B row stride
  __shared__ float As32[2][BM][LW];
  __shared__ float Bs32[2][BN][LW];

  const int tid = threadIdx.x;
  const int lane = tid & 31, wid = tid >> 5;
  const int waveM = wid & 3, waveN = wid >> 2;
  const int m0 = blockIdx.y * BM;
  const int n0 = blockIdx.x * BN;
  const int l15 = lane & 15;
  const int kh8  = (lane >> 4) * 8;
  const int kh16 = (lane >> 4) * 16;

  v8f acc[2][2] = {};

  // prologue: issue tiles 0 and 1 (2 TDM ops per tile-pair)
  if (wid == 0) {
    tdm_load_tile(&As32[0][0][0], A + (size_t)m0 * lda, BM, K, M - m0, lda);
    tdm_load_tile(&Bs32[0][0][0], W + (size_t)n0 * ldw, BN, K, N - n0, ldw);
    if (BK < K) {
      tdm_load_tile(&As32[1][0][0], A + (size_t)m0 * lda + BK, BM, K - BK, M - m0, lda);
      tdm_load_tile(&Bs32[1][0][0], W + (size_t)n0 * ldw + BK, BN, K - BK, N - n0, ldw);
    }
  }

  int cur = 0;
  for (int kk = 0; kk < K; kk += BK) {
    if (wid == 0) {
      if (kk + BK < K) __builtin_amdgcn_s_wait_tensorcnt(2);   // tile kk resident,
      else             __builtin_amdgcn_s_wait_tensorcnt(0);   // next pair may fly
    }
    __syncthreads();

    BF16Frag afr[2], bfr[2];
    #pragma unroll
    for (int mi = 0; mi < 2; ++mi) {
      int row = waveM * 32 + mi * 16 + l15;
      const float4* ap  = (const float4*)&As32[cur][row][kh16];
      const float4* ap2 = (const float4*)&As32[cur][row][kh16 + 8];
      float4 a0 = ap[0],  a1 = ap[1];
      float4 a2 = ap2[0], a3 = ap2[1];
      afr[mi].w[0] = pk2bf(a0.x, a0.y); afr[mi].w[1] = pk2bf(a0.z, a0.w);
      afr[mi].w[2] = pk2bf(a1.x, a1.y); afr[mi].w[3] = pk2bf(a1.z, a1.w);
      afr[mi].w[4] = pk2bf(a2.x, a2.y); afr[mi].w[5] = pk2bf(a2.z, a2.w);
      afr[mi].w[6] = pk2bf(a3.x, a3.y); afr[mi].w[7] = pk2bf(a3.z, a3.w);
    }
    #pragma unroll
    for (int ni = 0; ni < 2; ++ni) {
      int col = waveN * 32 + ni * 16 + l15;
      const float4* bp = (const float4*)&Bs32[cur][col][kh16];
      float4 b0 = bp[0], b1 = bp[1], b2 = bp[2], b3 = bp[3];
      bfr[ni].w[0] = pk2bf(b0.x, b0.y); bfr[ni].w[1] = pk2bf(b0.z, b0.w);
      bfr[ni].w[2] = pk2bf(b1.x, b1.y); bfr[ni].w[3] = pk2bf(b1.z, b1.w);
      bfr[ni].w[4] = pk2bf(b2.x, b2.y); bfr[ni].w[5] = pk2bf(b2.z, b2.w);
      bfr[ni].w[6] = pk2bf(b3.x, b3.y); bfr[ni].w[7] = pk2bf(b3.z, b3.w);
    }
    #pragma unroll
    for (int mi = 0; mi < 2; ++mi)
      #pragma unroll
      for (int ni = 0; ni < 2; ++ni)
        acc[mi][ni] = wmma_bf16(afr[mi], bfr[ni], acc[mi][ni]);

    __syncthreads();                              // all reads of `cur` done
    if (wid == 0 && kk + 2 * BK < K) {            // refill freed buffer (tile kk+2)
      tdm_load_tile(&As32[cur][0][0], A + (size_t)m0 * lda + kk + 2 * BK,
                    BM, K - kk - 2 * BK, M - m0, lda);
      tdm_load_tile(&Bs32[cur][0][0], W + (size_t)n0 * ldw + kk + 2 * BK,
                    BN, K - kk - 2 * BK, N - n0, ldw);
    }
    cur ^= 1;
  }

  // epilogue
  #pragma unroll
  for (int mi = 0; mi < 2; ++mi) {
    #pragma unroll
    for (int ni = 0; ni < 2; ++ni) {
      int col = n0 + waveN * 32 + ni * 16 + l15;
      if (col < N) {
        float bval = BIAS ? bias[col] : 0.f;
        #pragma unroll
        for (int r = 0; r < 8; ++r) {
          int row = m0 + waveM * 32 + mi * 16 + ((lane < 16) ? r : r + 8);
          float v = acc[mi][ni][r] + bval;
          if (GELU) v = gelu_exact(v);
          if (RSC)  v *= rscale[(size_t)row * rs_stride];
          if (SPLITQ) {
            // scatter into q/k/v [b,h,t,dh] (three contiguous TOK*DIM buffers)
            int which = col / DIM;
            int d = col - which * DIM;
            int hh = d >> 6, dd = d & 63;
            int bb = row >> 9, tq = row & (SEQ - 1);
            size_t dst = (size_t)which * (TOK * DIM) +
                         ((size_t)(bb * NHEAD + hh) * SEQ + tq) * DHEAD + dd;
            C[dst] = v;
          } else {
            float* cp = C + (size_t)row * ldc + col;
            if (ACC) *cp += v; else *cp = v;
          }
        }
      }
    }
  }
}

// ---------------- attention: 16 query rows / block ----------------
__global__ void __launch_bounds__(128) attention_kernel(
    const float* __restrict__ q, const float* __restrict__ k,
    const float* __restrict__ v, float* __restrict__ out) {
  constexpr int QW = 36;                         // uints per Q row (32+4)
  constexpr int SP = SEQ + 8;                    // padded score row
  __shared__ unsigned int   Qs[16][QW];
  __shared__ float          Ss[16][SP];
  __shared__ unsigned short Vst[DHEAD][40];      // transposed V slab (bf16)

  const int tid = threadIdx.x;
  const int lane = tid & 31, wid = tid >> 5;     // 4 waves
  const int qb = blockIdx.x;                     // 0..31
  const int bh = blockIdx.y;                     // 0..23
  const int bb = bh / NHEAD, hh = bh % NHEAD;
  const int q0 = qb * 16;
  const int l15 = lane & 15;
  const int kh8  = (lane >> 4) * 8;
  const int kh16 = (lane >> 4) * 16;
  const int khU  = (lane >> 4) * 4;

  const float* Qg = q + (size_t)bh * SEQ * DHEAD;
  const float* Kg = k + (size_t)bh * SEQ * DHEAD;
  const float* Vg = v + (size_t)bh * SEQ * DHEAD;

  // stage Q tile as bf16 (packed stores)
  for (int i = tid; i < 16 * 16; i += 128) {     // 256 float4 slots
    int r = i >> 4, c4 = (i & 15) * 4;
    float4 f = *(const float4*)(Qg + (size_t)(q0 + r) * DHEAD + c4);
    Qs[r][(c4 >> 1)]     = pk2bf(f.x, f.y);
    Qs[r][(c4 >> 1) + 1] = pk2bf(f.z, f.w);
  }
  __syncthreads();

  // phase 1: S = QK^T/8 + causal mask; wave w handles keys [w*128, w*128+128)
  for (int ns = 0; ns < 8; ++ns) {
    int n0 = wid * 128 + ns * 16;
    v8f acc = {};
    #pragma unroll
    for (int kb = 0; kb < DHEAD; kb += 32) {
      BF16Frag af, bf;
      af.q[0] = *(const uint4*)&Qs[l15][(kb >> 1) + kh8];
      af.q[1] = *(const uint4*)&Qs[l15][(kb >> 1) + 4 + kh8];
      const float4* kp = (const float4*)(Kg + (size_t)(n0 + l15) * DHEAD + kb + kh16);
      float4 k0 = kp[0], k1 = kp[1], k2 = kp[2], k3 = kp[3];
      bf.w[0] = pk2bf(k0.x, k0.y); bf.w[1] = pk2bf(k0.z, k0.w);
      bf.w[2] = pk2bf(k1.x, k1.y); bf.w[3] = pk2bf(k1.z, k1.w);
      bf.w[4] = pk2bf(k2.x, k2.y); bf.w[5] = pk2bf(k2.z, k2.w);
      bf.w[6] = pk2bf(k3.x, k3.y); bf.w[7] = pk2bf(k3.z, k3.w);
      acc = wmma_bf16(af, bf, acc);
    }
    #pragma unroll
    for (int r = 0; r < 8; ++r) {
      int row = (lane < 16) ? r : r + 8;
      int col = n0 + l15;
      float sv = acc[r] * 0.125f;                // 1/sqrt(64)
      if (col > q0 + row) sv = -1e30f;           // causal mask
      Ss[row][col] = sv;
    }
  }
  __syncthreads();

  // phase 2: row softmax; wave w handles rows [w*4, w*4+4)
  for (int rr = 0; rr < 4; ++rr) {
    int row = wid * 4 + rr;
    float m = -1e30f;
    for (int j = lane; j < SEQ; j += 32) m = fmaxf(m, Ss[row][j]);
    #pragma unroll
    for (int off = 16; off; off >>= 1) m = fmaxf(m, __shfl_xor(m, off));
    float s = 0.f;
    for (int j = lane; j < SEQ; j += 32) {
      float e = __expf(Ss[row][j] - m); Ss[row][j] = e; s += e;
    }
    #pragma unroll
    for (int off = 16; off; off >>= 1) s += __shfl_xor(s, off);
    float inv = 1.f / s;
    for (int j = lane; j < SEQ; j += 32) Ss[row][j] *= inv;
  }

  // phase 3: O = P V; wave w handles dh columns [w*16, w*16+16)
  const int n0 = wid * 16;
  v8f acc = {};
  for (int kb = 0; kb < SEQ; kb += 32) {
    __syncthreads();                             // protect Vst reuse (and Ss on 1st iter)
    // stage transposed V slab: Vst[dh][k] = V[kb+k][dh]
    #pragma unroll
    for (int j = 0; j < 4; ++j) {
      int i = tid + j * 128; int kr = i >> 4, c4 = (i & 15) * 4;
      float4 f = *(const float4*)(Vg + (size_t)(kb + kr) * DHEAD + c4);
      Vst[c4 + 0][kr] = f2bfb(f.x); Vst[c4 + 1][kr] = f2bfb(f.y);
      Vst[c4 + 2][kr] = f2bfb(f.z); Vst[c4 + 3][kr] = f2bfb(f.w);
    }
    __syncthreads();

    BF16Frag af, bf;
    #pragma unroll
    for (int i2 = 0; i2 < 4; ++i2) {
      float2 p0 = *(const float2*)&Ss[l15][kb + kh8 + 2 * i2];
      float2 p1 = *(const float2*)&Ss[l15][kb + 16 + kh8 + 2 * i2];
      af.w[i2]     = pk2bf(p0.x, p0.y);
      af.w[4 + i2] = pk2bf(p1.x, p1.y);
    }
    int col = n0 + l15;
    bf.q[0] = *(const uint4*)&Vst[col][kh8];
    bf.q[1] = *(const uint4*)&Vst[col][16 + kh8];
    acc = wmma_bf16(af, bf, acc);
  }
  #pragma unroll
  for (int r = 0; r < 8; ++r) {
    int row = (lane < 16) ? r : r + 8;
    int col = n0 + l15;
    out[((size_t)(bb * SEQ + q0 + row) * DIM) + hh * DHEAD + col] = acc[r];
  }
}

// ---------------- router: softmax over 8 experts, top-2, renormalized gates ----
__global__ void __launch_bounds__(256) router_kernel(const float* __restrict__ xn,
                                                     const float* __restrict__ gw,
                                                     float* __restrict__ cw) {
  const int tok = blockIdx.x;
  const int tid = threadIdx.x;
  const int lane = tid & 31, wid = tid >> 5;     // wid == expert
  __shared__ float gl[NEXP];

  float s = 0.f;
  for (int d = lane; d < DIM; d += 32)
    s += xn[(size_t)tok * DIM + d] * gw[(size_t)wid * DIM + d];
  #pragma unroll
  for (int off = 16; off; off >>= 1) s += __shfl_xor(s, off);
  if (lane == 0) gl[wid] = s;
  __syncthreads();

  if (tid == 0) {
    float m = gl[0];
    #pragma unroll
    for (int j = 1; j < NEXP; ++j) m = fmaxf(m, gl[j]);
    float e[NEXP];
    #pragma unroll
    for (int j = 0; j < NEXP; ++j) e[j] = __expf(gl[j] - m);
    int i1 = 0;
    #pragma unroll
    for (int j = 1; j < NEXP; ++j) if (e[j] > e[i1]) i1 = j;
    int i2 = (i1 == 0) ? 1 : 0;
    #pragma unroll
    for (int j = 0; j < NEXP; ++j) if (j != i1 && e[j] > e[i2]) i2 = j;
    float wsum = e[i1] + e[i2];
    #pragma unroll
    for (int j = 0; j < NEXP; ++j) cw[(size_t)tok * NEXP + j] = 0.f;
    cw[(size_t)tok * NEXP + i1] = e[i1] / wsum;
    cw[(size_t)tok * NEXP + i2] = e[i2] / wsum;
  }
}

// ---------------- host-side orchestration ----------------
extern "C" void kernel_launch(void* const* d_in, const int* in_sizes, int n_in,
                              void* d_out, int out_size, void* d_ws, size_t ws_size,
                              hipStream_t stream) {
  const int*   x    = (const int*)  d_in[0];
  const float* emb  = (const float*)d_in[1];
  const float* wqkv = (const float*)d_in[2];
  const float* bqkv = (const float*)d_in[3];
  const float* wo   = (const float*)d_in[4];
  const float* bo   = (const float*)d_in[5];
  const float* ln1g = (const float*)d_in[6];
  const float* ln1b = (const float*)d_in[7];
  const float* ln2g = (const float*)d_in[8];
  const float* ln2b = (const float*)d_in[9];
  const float* sw1  = (const float*)d_in[10];
  const float* sb1  = (const float*)d_in[11];
  const float* sw2  = (const float*)d_in[12];
  const float* sb2  = (const float*)d_in[13];
  const float* gw   = (const float*)d_in[14];
  const float* mw1  = (const float*)d_in[15];
  const float* mb1  = (const float*)d_in[16];
  const float* mw2  = (const float*)d_in[17];
  const float* mb2  = (const float*)d_in[18];
  const float* lnfg = (const float*)d_in[19];
  const float* lnfb = (const float*)d_in[20];
  float* out = (float*)d_out;

  float* ws = (float*)d_ws;
  float* h     = ws; ws += (size_t)TOK * DIM;
  float* xn    = ws; ws += (size_t)TOK * DIM;
  float* qh    = ws; ws += (size_t)TOK * DIM;   // q/k/v contiguous for fused split
  float* kh    = ws; ws += (size_t)TOK * DIM;
  float* vh    = ws; ws += (size_t)TOK * DIM;
  float* attno = ws; ws += (size_t)TOK * DIM;
  float* t1    = ws; ws += (size_t)TOK * HID;
  float* cw    = ws; ws += (size_t)TOK * NEXP;

  const dim3 blk256(256), blk128(128);
  const dim3 gQKV(3 * DIM / 64, TOK / 128);     // N=2304
  const dim3 gD  (DIM / 64,     TOK / 128);     // N=768
  const dim3 gH  (HID / 64,     TOK / 128);     // N=3072
  const dim3 gV  ((VOCAB + 63) / 64, TOK / 128);
  const dim3 gAtt(SEQ / 16, BATCH * NHEAD);

  embed_kernel<<<TOK, blk256, 0, stream>>>(x, emb, h);

  int std_i = 0, moe_i = 0;
  for (int l = 0; l < NLAYER; ++l) {
    // attention
    ln_kernel<<<TOK, blk256, 0, stream>>>(h, ln1g + l * DIM, ln1b + l * DIM, xn);
    gemm_bf16_kernel<1,0,0,0,1><<<gQKV, blk256, 0, stream>>>(      // fused head split
        xn, DIM, wqkv + (size_t)l * 3 * DIM * DIM, DIM, bqkv + l * 3 * DIM,
        nullptr, 0, qh, DIM, TOK, 3 * DIM, DIM);
    attention_kernel<<<gAtt, blk128, 0, stream>>>(qh, kh, vh, attno);
    gemm_bf16_kernel<1,0,1,0,0><<<gD, blk256, 0, stream>>>(
        attno, DIM, wo + (size_t)l * DIM * DIM, DIM, bo + l * DIM,
        nullptr, 0, h, DIM, TOK, DIM, DIM);

    // FFN
    ln_kernel<<<TOK, blk256, 0, stream>>>(h, ln2g + l * DIM, ln2b + l * DIM, xn);
    if (l == 1 || l == 3) {
      router_kernel<<<TOK, blk256, 0, stream>>>(xn, gw + (size_t)moe_i * NEXP * DIM, cw);
      for (int e = 0; e < NEXP; ++e) {
        const float* w1 = mw1 + ((size_t)moe_i * NEXP + e) * HID * DIM;
        const float* b1 = mb1 + ((size_t)moe_i * NEXP + e) * HID;
        const float* w2 = mw2 + ((size_t)moe_i * NEXP + e) * DIM * HID;
        const float* b2 = mb2 + ((size_t)moe_i * NEXP + e) * DIM;
        gemm_bf16_kernel<1,1,0,0,0><<<gH, blk256, 0, stream>>>(
            xn, DIM, w1, DIM, b1, nullptr, 0, t1, HID, TOK, HID, DIM);
        gemm_bf16_kernel<1,0,1,1,0><<<gD, blk256, 0, stream>>>(
            t1, HID, w2, HID, b2, cw + e, NEXP, h, DIM, TOK, DIM, HID);
      }
      moe_i++;
    } else {
      gemm_bf16_kernel<1,1,0,0,0><<<gH, blk256, 0, stream>>>(
          xn, DIM, sw1 + (size_t)std_i * HID * DIM, DIM, sb1 + std_i * HID,
          nullptr, 0, t1, HID, TOK, HID, DIM);
      gemm_bf16_kernel<1,0,1,0,0><<<gD, blk256, 0, stream>>>(
          t1, HID, sw2 + (size_t)std_i * DIM * HID, HID, sb2 + std_i * DIM,
          nullptr, 0, h, DIM, TOK, DIM, HID);
      std_i++;
    }
  }

  // final LN + tied-embedding logits
  ln_kernel<<<TOK, blk256, 0, stream>>>(h, lnfg, lnfb, xn);
  gemm_bf16_kernel<0,0,0,0,0><<<gV, blk256, 0, stream>>>(
      xn, DIM, emb, DIM, nullptr, nullptr, 0, out, VOCAB, TOK, VOCAB, DIM);
}
